// MultiHeadAttention_5480378270305
// MI455X (gfx1250) — matrix operand order repaired and checked by reading, not compiled, not run on hardware
//
#include <hip/hip_runtime.h>
#include <hip/hip_bf16.h>
#include <math.h>

// ---------------- types ----------------
typedef __bf16 bf16_t;
typedef __attribute__((ext_vector_type(16))) __bf16 v16bf;
typedef __attribute__((ext_vector_type(8)))  __bf16 v8bf;
typedef __attribute__((ext_vector_type(8)))  float  v8f;
typedef __attribute__((ext_vector_type(4)))  int    v4i;

// ---------------- problem constants ----------------
static constexpr int Bn  = 2;
static constexpr int Sn  = 2048;
static constexpr int Dn  = 1024;
static constexpr int Hn  = 16;
static constexpr int Mn  = Bn * Sn;     // 4096 rows
static constexpr int N3  = 3 * Dn;      // 3072

// CDNA5 async memory->LDS path (ASYNCcnt), guarded so we keep a fallback.
#if defined(__has_builtin)
#  if __has_builtin(__builtin_amdgcn_global_load_async_to_lds_b128) && \
      __has_builtin(__builtin_amdgcn_s_wait_asynccnt)
#    define USE_ASYNC_LDS 1
#  endif
#endif
#ifndef USE_ASYNC_LDS
#  define USE_ASYNC_LDS 0
#endif

#if USE_ASYNC_LDS
typedef __attribute__((address_space(1))) v4i gv4i;   // global int4
typedef __attribute__((address_space(3))) v4i lv4i;   // LDS int4
#endif

// ds_swizzle XOR butterfly (group-of-32 mode: offset = {0, xor[14:10], or[9:5], and[4:0]})
template <int XM>
__device__ __forceinline__ float swz_xor(float x) {
  int v = __builtin_amdgcn_ds_swizzle(__float_as_int(x), (XM << 10) | 0x1f);
  return __int_as_float(v);
}

// Batched butterfly stages: all swizzles into distinct temps first so the
// scheduler can clause the DS ops instead of wait-per-shuffle.
template <int XM>
__device__ __forceinline__ void max_stage(float (&a)[8]) {
  float t[8];
#pragma unroll
  for (int r = 0; r < 8; ++r) t[r] = swz_xor<XM>(a[r]);
#pragma unroll
  for (int r = 0; r < 8; ++r) a[r] = fmaxf(a[r], t[r]);
}
template <int XM>
__device__ __forceinline__ void add_stage(float (&a)[8]) {
  float t[8];
#pragma unroll
  for (int r = 0; r < 8; ++r) t[r] = swz_xor<XM>(a[r]);
#pragma unroll
  for (int r = 0; r < 8; ++r) a[r] += t[r];
}

// =====================================================================
// Convert f32 -> bf16 (flat)
// =====================================================================
__global__ void k_cvt(const float* __restrict__ src, bf16_t* __restrict__ dst, int n) {
  int i = blockIdx.x * 256 + threadIdx.x;
  if (i < n) dst[i] = (bf16_t)src[i];
}

// Convert + transpose: src[rows][cols] f32 -> dst[cols][rows] bf16
__global__ void k_cvt_T(const float* __restrict__ src, bf16_t* __restrict__ dst,
                        int rows, int cols) {
  int i = blockIdx.x * 256 + threadIdx.x;
  if (i < rows * cols) {
    int c = i / rows;
    int r = i % rows;
    dst[i] = (bf16_t)src[(size_t)r * cols + c];
  }
}

// =====================================================================
// WMMA GEMM: C[M][N] = A[M][K](bf16,row-major) x BT[N][K](bf16) + bias
// One wave computes a 32x64 tile (2 A-tiles x 4 B-tiles, 8 accumulators),
// 8 waves / block. Each B operand is reused by two WMMAs.
// =====================================================================
template <typename OutT>
__global__ __launch_bounds__(256, 1) void k_gemm(
                       const bf16_t* __restrict__ A,
                       const bf16_t* __restrict__ BT,
                       const float*  __restrict__ bias,
                       OutT* __restrict__ C,
                       int M, int N, int K) {
  const int lane = threadIdx.x & 31;
  const int wave = threadIdx.x >> 5;
  const int wid  = blockIdx.x * 8 + wave;
  const int nt64 = N / 64;
  const int m0 = (wid / nt64) * 32;
  const int n0 = (wid % nt64) * 64;
  if (m0 >= M) return;

  const int ksub = (lane < 16) ? 0 : 16;   // A-layout K sub-offset
  const int bsub = (lane < 16) ? 0 : 16;   // B-layout K half
  const bf16_t* arow0 = A + (size_t)(m0 + (lane & 15)) * K;
  const bf16_t* arow1 = arow0 + (size_t)16 * K;

  v8f acc[2][4] = {};
#pragma unroll 2
  for (int k0 = 0; k0 < K; k0 += 32) {
    v16bf a0, a1;
    {
      v8bf lo = *(const v8bf*)(arow0 + k0 + ksub);
      v8bf hi = *(const v8bf*)(arow0 + k0 + 8 + ksub);
#pragma unroll
      for (int i = 0; i < 8; ++i) { a0[i] = lo[i]; a0[i + 8] = hi[i]; }
    }
    {
      v8bf lo = *(const v8bf*)(arow1 + k0 + ksub);
      v8bf hi = *(const v8bf*)(arow1 + k0 + 8 + ksub);
#pragma unroll
      for (int i = 0; i < 8; ++i) { a1[i] = lo[i]; a1[i + 8] = hi[i]; }
    }
    // load all 4 B operands first (one clause), then 8 back-to-back WMMAs
    v16bf bm[4];
#pragma unroll
    for (int t = 0; t < 4; ++t)
      bm[t] = *(const v16bf*)(BT + (size_t)(n0 + t * 16 + (lane & 15)) * K + k0 + bsub);
#pragma unroll
    for (int t = 0; t < 4; ++t) {
      acc[0][t] = __builtin_amdgcn_wmma_f32_16x16x32_bf16(
          false, a0, false, bm[t], (short)0, acc[0][t], false, false);
      acc[1][t] = __builtin_amdgcn_wmma_f32_16x16x32_bf16(
          false, a1, false, bm[t], (short)0, acc[1][t], false, false);
    }
  }

  // epilogue: C layout row = base + r + (lane<16 ? 0 : 8)
#pragma unroll
  for (int g = 0; g < 2; ++g) {
    const int row0 = m0 + g * 16 + ((lane < 16) ? 0 : 8);
#pragma unroll
    for (int t = 0; t < 4; ++t) {
      const int col = n0 + t * 16 + (lane & 15);
      const float bv = bias[col];
#pragma unroll
      for (int r = 0; r < 8; ++r)
        C[(size_t)(row0 + r) * N + col] = (OutT)(acc[g][t][r] + bv);
    }
  }
}

// =====================================================================
// Flash attention. qkv bf16 [B*S][3D], row layout per head h:
// h*192+{0..63:q, 64..127:k, 128..191:v}. One wave = 32 query rows
// (two 16-row A-groups sharing every K/V tile); 4 waves / block =
// 128 query rows per (b,h) block.
// =====================================================================
__global__ __launch_bounds__(128, 1) void k_attn(
                       const bf16_t* __restrict__ qkv,
                       const int*    __restrict__ counts,
                       bf16_t* __restrict__ vals) {
  __shared__ __align__(16) bf16_t ldsV[4][32 * 64]; // V tile per wave
  __shared__ __align__(16) bf16_t ldsP[4][32 * 32]; // P transpose buffer (2 groups)

  const int lane = threadIdx.x & 31;
  const int wave = threadIdx.x >> 5;
  const int qb   = blockIdx.x % (Sn / 128);
  const int bh   = blockIdx.x / (Sn / 128);
  const int h    = bh % Hn;
  const int b    = bh / Hn;
  const int q0   = qb * 128 + wave * 32;
  const int valid = Sn - counts[b];          // keys >= valid are masked

  const size_t rs = (size_t)N3;              // 3072 row stride
  const bf16_t* base = qkv + (size_t)b * Sn * rs;
  const int qoff = h * 192;
  const int koff = h * 192 + 64;
  const int voff = h * 192 + 128;

  const int ksub = (lane < 16) ? 0 : 16;

  // ---- Q tiles (2 row groups), A layout, two K-chunks (d0 = 0, 32) ----
  v16bf qA[2][2];
#pragma unroll
  for (int g = 0; g < 2; ++g) {
    const int arow = q0 + g * 16 + (lane & 15);
#pragma unroll
    for (int c = 0; c < 2; ++c) {
      const bf16_t* p = base + (size_t)arow * rs + qoff + 32 * c;
      v8bf lo = *(const v8bf*)(p + ksub);
      v8bf hi = *(const v8bf*)(p + 8 + ksub);
#pragma unroll
      for (int i = 0; i < 8; ++i) { qA[g][c][i] = lo[i]; qA[g][c][i + 8] = hi[i]; }
    }
  }

  float mrow[2][8], lrow[2][8];
  v8f o[2][4] = {};              // output accumulators, [group][d-chunk]
#pragma unroll
  for (int g = 0; g < 2; ++g)
#pragma unroll
    for (int r = 0; r < 8; ++r) { mrow[g][r] = -__builtin_inff(); lrow[g][r] = 0.f; }

  const int   nkt   = (valid + 31) / 32;
  const float scale = 0.125f;    // 1/sqrt(64)

  for (int kt = 0; kt < nkt; ++kt) {
    const int kb = kt * 32;

    if (kt + 1 < nkt)            // prefetch next K tile -> global_prefetch_b8
      __builtin_prefetch(base + (size_t)(kb + 32 + lane) * rs + koff, 0, 0);

    // ---- V tile -> LDS (32x64 bf16) via async DMA, lane owns one key row ----
    {
      const bf16_t* vp = base + (size_t)(kb + lane) * rs + voff;
      bf16_t* dst = &ldsV[wave][lane * 64];
#if USE_ASYNC_LDS
      gv4i* gsrc = (gv4i*)vp;
      lv4i* ldst = (lv4i*)dst;
#pragma unroll
      for (int i = 0; i < 8; ++i)
        __builtin_amdgcn_global_load_async_to_lds_b128(gsrc + i, ldst + i, 0, 0);
#else
#pragma unroll
      for (int i = 0; i < 8; ++i)
        *(v8bf*)(dst + i * 8) = *(const v8bf*)(vp + i * 8);
#endif
    }

    // ---- scores: both groups share the K operands ----
    v16bf kB[2][2];
#pragma unroll
    for (int t = 0; t < 2; ++t)
#pragma unroll
      for (int c = 0; c < 2; ++c)
        kB[t][c] = *(const v16bf*)(base + (size_t)(kb + 16 * t + (lane & 15)) * rs +
                                   koff + 32 * c + ((lane < 16) ? 0 : 16));
    v8f s[2][2] = {};
#pragma unroll
    for (int g = 0; g < 2; ++g)
#pragma unroll
      for (int t = 0; t < 2; ++t)
#pragma unroll
        for (int c = 0; c < 2; ++c)
          s[g][t] = __builtin_amdgcn_wmma_f32_16x16x32_bf16(
              false, qA[g][c], false, kB[t][c], (short)0, s[g][t], false, false);

    // ---- per-group masked online softmax ----
    const int  key0 = kb + (lane & 15);
    const bool m0k  = key0 >= valid;
    const bool m1k  = (key0 + 16) >= valid;

#pragma unroll
    for (int g = 0; g < 2; ++g) {
#pragma unroll
      for (int r = 0; r < 8; ++r) {
        s[g][0][r] = m0k ? -__builtin_inff() : s[g][0][r] * scale;
        s[g][1][r] = m1k ? -__builtin_inff() : s[g][1][r] * scale;
      }
      float mx[8];
#pragma unroll
      for (int r = 0; r < 8; ++r) mx[r] = fmaxf(s[g][0][r], s[g][1][r]);
      max_stage<1>(mx); max_stage<2>(mx); max_stage<4>(mx); max_stage<8>(mx);

      // exp + immediate P store (keeps p0/p1 live range minimal)
      float alpha[8], ps[8];
#pragma unroll
      for (int r = 0; r < 8; ++r) {
        const float mn = fmaxf(mrow[g][r], mx[r]);
        const float p0 = __expf(s[g][0][r] - mn);   // exp(-inf)=0 for masked keys
        const float p1 = __expf(s[g][1][r] - mn);
        alpha[r] = __expf(mrow[g][r] - mn);
        mrow[g][r] = mn;
        ps[r]    = p0 + p1;
        const int prow = g * 16 + r + ((lane < 16) ? 0 : 8);
        ldsP[wave][prow * 32 + (lane & 15)]      = (bf16_t)p0;
        ldsP[wave][prow * 32 + 16 + (lane & 15)] = (bf16_t)p1;
      }
      add_stage<1>(ps); add_stage<2>(ps); add_stage<4>(ps); add_stage<8>(ps);

#pragma unroll
      for (int r = 0; r < 8; ++r)
        lrow[g][r] = lrow[g][r] * alpha[r] + ps[r];
      // rescale running output for this group
#pragma unroll
      for (int c = 0; c < 4; ++c)
#pragma unroll
        for (int r = 0; r < 8; ++r) o[g][c][r] *= alpha[r];
    }

    // ---- P back from LDS in A layout (both groups) ----
    v16bf pA[2];
#pragma unroll
    for (int g = 0; g < 2; ++g) {
      const bf16_t* pp = &ldsP[wave][(g * 16 + (lane & 15)) * 32];
      v8bf lo = *(const v8bf*)(pp + ksub);
      v8bf hi = *(const v8bf*)(pp + 8 + ksub);
#pragma unroll
      for (int i = 0; i < 8; ++i) { pA[g][i] = lo[i]; pA[g][i + 8] = hi[i]; }
    }

#if USE_ASYNC_LDS
    __builtin_amdgcn_s_wait_asynccnt(0);   // V tile resident in LDS
#endif

    // ---- O += P x V: gather the 4 B chunks once, reuse for both groups ----
    const int kbase = (lane < 16) ? 0 : 16;
    v16bf vB[4];
#pragma unroll
    for (int c = 0; c < 4; ++c) {
      const int dcol = c * 16 + (lane & 15);
#pragma unroll
      for (int i = 0; i < 16; ++i)
        vB[c][i] = ldsV[wave][(kbase + i) * 64 + dcol];
    }
#pragma unroll
    for (int c = 0; c < 4; ++c)
#pragma unroll
      for (int g = 0; g < 2; ++g)
        o[g][c] = __builtin_amdgcn_wmma_f32_16x16x32_bf16(
            false, pA[g], false, vB[c], (short)0, o[g][c], false, false);
  }

  // ---- finalize: divide by l, write bf16 vals[b,s, h*64+d] ----
  bf16_t* outp = vals + (size_t)b * Sn * Dn;
#pragma unroll
  for (int g = 0; g < 2; ++g) {
    const int row0 = q0 + g * 16 + ((lane < 16) ? 0 : 8);
#pragma unroll
    for (int c = 0; c < 4; ++c) {
      const int col = h * 64 + c * 16 + (lane & 15);
#pragma unroll
      for (int r = 0; r < 8; ++r) {
        const float v = o[g][c][r] / lrow[g][r];
        outp[(size_t)(row0 + r) * Dn + col] = (bf16_t)v;
      }
    }
  }
}

// =====================================================================
// launch
// =====================================================================
extern "C" void kernel_launch(void* const* d_in, const int* in_sizes, int n_in,
                              void* d_out, int out_size, void* d_ws, size_t ws_size,
                              hipStream_t stream) {
  const float* x      = (const float*)d_in[0];
  const int*   counts = (const int*)  d_in[1];
  const float* Wqkv   = (const float*)d_in[2];
  const float* bqkv   = (const float*)d_in[3];
  const float* Wo     = (const float*)d_in[4];
  const float* bo     = (const float*)d_in[5];
  float* out = (float*)d_out;

  char* ws = (char*)d_ws;
  bf16_t* xb    = (bf16_t*)ws; ws += (size_t)Mn * Dn * 2;     //  8.4 MB
  bf16_t* WqkvT = (bf16_t*)ws; ws += (size_t)N3 * Dn * 2;     //  6.3 MB
  bf16_t* WoT   = (bf16_t*)ws; ws += (size_t)Dn * Dn * 2;     //  2.1 MB
  bf16_t* qkvb  = (bf16_t*)ws; ws += (size_t)Mn * N3 * 2;     // 25.2 MB
  bf16_t* valsb = (bf16_t*)ws;                                //  8.4 MB

  // 1) precision conversion (+ weight transposes so B operands are contiguous)
  k_cvt  <<<(Mn * Dn + 255) / 256, 256, 0, stream>>>(x, xb, Mn * Dn);
  k_cvt_T<<<(Dn * N3 + 255) / 256, 256, 0, stream>>>(Wqkv, WqkvT, Dn, N3);
  k_cvt_T<<<(Dn * Dn + 255) / 256, 256, 0, stream>>>(Wo, WoT, Dn, Dn);

  // 2) QKV projection: [4096,1024] x [1024,3072] + bias -> bf16 qkv
  k_gemm<bf16_t><<<(Mn / 32) * (N3 / 64) / 8, 256, 0, stream>>>(
      xb, WqkvT, bqkv, qkvb, Mn, N3, Dn);

  // 3) masked flash attention -> bf16 vals [B*S][D]
  k_attn<<<Bn * Hn * (Sn / 128), 128, 0, stream>>>(qkvb, counts, valsb);

  // 4) output projection: [4096,1024] x [1024,1024] + bias -> f32 out
  k_gemm<float><<<(Mn / 32) * (Dn / 64) / 8, 256, 0, stream>>>(
      valsb, WoT, bo, out, Mn, Dn, Dn);
}
